// SPELL_59399397704350
// MI455X (gfx1250) — matrix operand order repaired and checked by reading, not compiled, not run on hardware
//
#include <hip/hip_runtime.h>
#include <hip/hip_bf16.h>

typedef __attribute__((ext_vector_type(16))) __bf16 v16bf;
typedef __attribute__((ext_vector_type(8)))  float  v8f;

#define C_DIM 128

__device__ inline v8f wmma_bf16(v16bf a, v16bf b, v8f c) {
    // D = A(16x32 bf16) * B(32x16 bf16) + C(16x16 f32)
    return __builtin_amdgcn_wmma_f32_16x16x32_bf16(false, a, false, b, (short)0, c, false, false);
}

__device__ inline v8f vzero8() { v8f z = {0.f,0.f,0.f,0.f,0.f,0.f,0.f,0.f}; return z; }

// K index for 16-bit A fragment element e, lane k-group kg(=0/8):
// contiguous: K = 2*kg + e (lane half h holds K = 16h..16h+15)
__device__ inline int a_kidx(int e, int kg) {
    return 2 * kg + e;
}

// ---------------------------------------------------------------------------
// Input projection: h0 = x[:, :1024]@W011 + x[:,1024:]@W012 + (b011+b012)
// [N,2048]@[2048,128]; 128 rows/block (8 waves x 16 rows).
// Weight k-slices staged in LDS in *fragment-major bf16* layout:
//   Bf[buf][tile][lane][16 contiguous K values]  -> one 32B LDS read per frag.
// Double-buffered + software-pipelined global loads.
// ---------------------------------------------------------------------------
__global__ void __launch_bounds__(256)
k_inproj(const float* __restrict__ x, const float* __restrict__ W1,
         const float* __restrict__ W2, const float* __restrict__ b1,
         const float* __restrict__ b2, float* __restrict__ out, int n)
{
    __shared__ __attribute__((aligned(32))) __bf16 Bf[2][8 * 32 * 16];  // 2 x 8KB
    const int tid  = threadIdx.x;
    const int lane = tid & 31;
    const int wave = tid >> 5;
    const int rowBase = blockIdx.x * 128 + wave * 16;
    const int m   = lane & 15;
    const int row = rowBase + m;
    const bool ok = row < n;

    // staging role: thread owns fragment chunk (ts, ls): 16 bf16 values
    const int ts = tid >> 5;
    const int ls = tid & 31;
    const int ns = ts * 16 + (ls & 15);      // weight column
    const int kr = (ls >> 4) * 16;           // base K within 32-slice

    auto load_wslice = [&](int k0, float* r) {
        #pragma unroll
        for (int v = 0; v < 16; ++v) {
            int gk = k0 + kr + v;
            r[v] = (gk < 1024) ? W1[gk * C_DIM + ns] : W2[(gk - 1024) * C_DIM + ns];
        }
    };
    auto store_wslice = [&](int buf, const float* r) {
        v16bf p;
        #pragma unroll
        for (int v = 0; v < 16; ++v) p[v] = (__bf16)r[v];
        *(v16bf*)&Bf[buf][(ts * 32 + ls) * 16] = p;
    };

    const int kgA = (lane >> 4) * 8;
    const float* xrow = x + (size_t)(ok ? row : 0) * 2048;
    auto load_a = [&](int k0, float* r) {
        #pragma unroll
        for (int e = 0; e < 16; ++e) {
            float v = xrow[k0 + a_kidx(e, kgA)];
            r[e] = ok ? v : 0.f;
        }
    };

    v8f acc[8];
    #pragma unroll
    for (int t = 0; t < 8; ++t) acc[t] = vzero8();

    float wreg[16], areg[16];
    load_wslice(0, wreg);
    store_wslice(0, wreg);
    load_a(0, areg);
    __syncthreads();

    for (int k0 = 0; k0 < 2048; k0 += 32) {
        const int  cur  = (k0 >> 5) & 1;
        const bool more = (k0 + 32) < 2048;
        float wnext[16], anext[16];
        if (more) {                               // issue next-slice loads early
            load_wslice(k0 + 32, wnext);
            load_a(k0 + 32, anext);
        }
        if (ok && (k0 + 64) < 2048)
            __builtin_prefetch(xrow + k0 + 64, 0, 1);   // global_prefetch

        v16bf a;
        #pragma unroll
        for (int e = 0; e < 16; ++e) a[e] = (__bf16)areg[e];

        #pragma unroll
        for (int t = 0; t < 8; ++t) {
            v16bf b = *(const v16bf*)&Bf[cur][(t * 32 + lane) * 16];
            acc[t] = wmma_bf16(a, b, acc[t]);
        }
        __syncthreads();
        if (more) store_wslice(cur ^ 1, wnext);
        __syncthreads();
        if (more) {
            #pragma unroll
            for (int e = 0; e < 16; ++e) areg[e] = anext[e];
        }
    }

    #pragma unroll
    for (int t = 0; t < 8; ++t) {
        int c = t * 16 + m;
        float bias = b1[c] + b2[c];
        #pragma unroll
        for (int v = 0; v < 8; ++v) {
            int r = rowBase + ((lane < 16) ? v : v + 8);
            if (r < n) out[(size_t)r * C_DIM + c] = acc[t][v] + bias;
        }
    }
}

// ---------------------------------------------------------------------------
// out = A1@W1 [+ (A2*1/max(deg,1))@W2] [+ bias];  all [N,128]@[128,128]
// Both weights staged up-front in fragment-major bf16 LDS (32KB each):
//   w[ kstep ][ tile ][ lane ][ 16 values ]  -> one 32B LDS read per WMMA.
// ---------------------------------------------------------------------------
__global__ void __launch_bounds__(256)
k_gemm128(const float* __restrict__ A1, const float* __restrict__ W1,
          const float* __restrict__ A2, const float* __restrict__ W2,
          const float* __restrict__ deg, const float* __restrict__ bias,
          float* __restrict__ out, int n)
{
    __shared__ __attribute__((aligned(32))) __bf16 w1s[4 * 8 * 32 * 16];  // 32KB
    __shared__ __attribute__((aligned(32))) __bf16 w2s[4 * 8 * 32 * 16];  // 32KB
    const int tid  = threadIdx.x;
    const int lane = tid & 31;
    const int wave = tid >> 5;

    for (int c = tid; c < 1024; c += 256) {   // 1024 chunks of 16 values
        int ks = c >> 8;
        int t  = (c >> 5) & 7;
        int l  = c & 31;
        int nn = t * 16 + (l & 15);
        int kb = ks * 32 + (l >> 4) * 16;
        v16bf p1, p2;
        #pragma unroll
        for (int v = 0; v < 16; ++v) {
            p1[v] = (__bf16)W1[(kb + v) * C_DIM + nn];
            p2[v] = W2 ? (__bf16)W2[(kb + v) * C_DIM + nn] : (__bf16)0.f;
        }
        *(v16bf*)&w1s[c * 16] = p1;
        *(v16bf*)&w2s[c * 16] = p2;
    }
    __syncthreads();

    const int rowBase = blockIdx.x * 128 + wave * 16;
    const int m   = lane & 15;
    const int kg  = (lane >> 4) * 8;
    const int row = rowBase + m;
    const bool ok = row < n;

    float scale = 1.f;
    if (A2 && ok) scale = 1.f / fmaxf(deg[row], 1.f);

    v8f acc[8];
    #pragma unroll
    for (int t = 0; t < 8; ++t) acc[t] = vzero8();

    const float* ar  = A1 + (size_t)(ok ? row : 0) * C_DIM;
    const float* a2r = A2 ? (A2 + (size_t)(ok ? row : 0) * C_DIM) : nullptr;

    #pragma unroll
    for (int ks = 0; ks < 4; ++ks) {
        const int k0 = ks * 32;
        v16bf a1, a2;
        #pragma unroll
        for (int e = 0; e < 16; ++e) {
            int kk = k0 + a_kidx(e, kg);
            a1[e] = (__bf16)(ok ? ar[kk] : 0.f);
            if (A2) a2[e] = (__bf16)(ok ? a2r[kk] * scale : 0.f);
        }
        #pragma unroll
        for (int t = 0; t < 8; ++t) {
            const int foff = ((ks * 8 + t) * 32 + lane) * 16;
            v16bf b1 = *(const v16bf*)&w1s[foff];
            acc[t] = wmma_bf16(a1, b1, acc[t]);
            if (A2) {
                v16bf b2 = *(const v16bf*)&w2s[foff];
                acc[t] = wmma_bf16(a2, b2, acc[t]);
            }
        }
    }

    #pragma unroll
    for (int t = 0; t < 8; ++t) {
        int c = t * 16 + m;
        float bv = bias ? bias[c] : 0.f;
        #pragma unroll
        for (int v = 0; v < 8; ++v) {
            int r = rowBase + ((lane < 16) ? v : v + 8);
            if (r < n) out[(size_t)r * C_DIM + c] = acc[t][v] + bv;
        }
    }
}

// ---------------------------------------------------------------------------
// Masked neighbor aggregation: agg[dst] += x[src]; deg[dst] += 1
// ---------------------------------------------------------------------------
__global__ void k_edge_agg(const float* __restrict__ x, const int* __restrict__ src,
                           const int* __restrict__ dst, const float* __restrict__ ea,
                           int mode, float* __restrict__ agg, float* __restrict__ deg, int ne)
{
    int g = blockIdx.x * blockDim.x + threadIdx.x;
    int e = g >> 2, part = g & 3;
    if (e >= ne) return;
    float a = ea[e];
    bool m = (mode == 2) || (mode == 0 ? (a <= 0.f) : (a >= 0.f));
    if (!m) return;
    int s = src[e], d = dst[e];
    const float* xs = x + (size_t)s * C_DIM + part * 32;
    float* ad = agg + (size_t)d * C_DIM + part * 32;
    #pragma unroll
    for (int j = 0; j < 32; ++j) atomicAdd(ad + j, xs[j]);
    if (part == 0) atomicAdd(deg + d, 1.f);
}

// ---------------------------------------------------------------------------
// GraphNorm: column sum / sumsq  ->  per-feature affine coeffs  ->  apply+GELU
// ---------------------------------------------------------------------------
__global__ void __launch_bounds__(256)
k_colstats(const float* __restrict__ x, float* __restrict__ stats, int n)
{
    __shared__ float ssum[256], ssq[256];
    int tid = threadIdx.x;
    int f = tid & 127, half = tid >> 7;
    int r0 = blockIdx.x * 64 + half * 32;
    float s = 0.f, q = 0.f;
    #pragma unroll 4
    for (int i = 0; i < 32; ++i) {
        int r = r0 + i;
        if (r < n) { float v = x[(size_t)r * C_DIM + f]; s += v; q += v * v; }
    }
    ssum[tid] = s; ssq[tid] = q;
    __syncthreads();
    if (tid < 128) {
        atomicAdd(&stats[f],       ssum[tid] + ssum[tid + 128]);
        atomicAdd(&stats[128 + f], ssq[tid]  + ssq[tid + 128]);
    }
}

__global__ void k_gncoef(const float* __restrict__ stats, const float* __restrict__ gamma,
                         const float* __restrict__ beta, const float* __restrict__ alpha,
                         float* __restrict__ coef, int n)
{
    int f = threadIdx.x;
    if (f >= 128) return;
    float invn = 1.f / (float)n;
    float mu   = stats[f] * invn;
    float ex2  = stats[128 + f] * invn;
    float al   = alpha[f];
    float var  = ex2 - mu * mu * al * (2.f - al);   // E[(x - al*mu)^2]
    float rs   = rsqrtf(var + 1e-5f);
    coef[f]       = gamma[f] * rs;
    coef[128 + f] = beta[f] - gamma[f] * al * mu * rs;
}

__global__ void k_gngelu(float* __restrict__ x, const float* __restrict__ coef, size_t total)
{
    size_t i = (size_t)blockIdx.x * blockDim.x + threadIdx.x;
    if (i >= total) return;
    int f = (int)(i & 127);
    float v = coef[f] * x[i] + coef[128 + f];
    x[i] = 0.5f * v * (1.f + erff(v * 0.70710678118654752f));  // exact GELU
}

// ---------------------------------------------------------------------------
// GATv2 edge passes. Virtual edges i>=E are self loops (always unmasked).
// ---------------------------------------------------------------------------
__device__ inline int f2ord(float f) { int b = __float_as_int(f); return b >= 0 ? b : (b ^ 0x7fffffff); }
__device__ inline float ord2f(int o) { int b = o >= 0 ? o : (o ^ 0x7fffffff); return __int_as_float(b); }

__global__ void k_gat_e(const float* __restrict__ l, const float* __restrict__ r,
                        const int* __restrict__ src, const int* __restrict__ dst,
                        const float* __restrict__ ea, int mode, const float* __restrict__ att,
                        float* __restrict__ ebuf, int* __restrict__ mx, int ne, int n)
{
    int g = blockIdx.x * blockDim.x + threadIdx.x;
    int i = g >> 1, h = g & 1;
    if (i >= ne + n) return;
    int s, d; bool m;
    if (i < ne) { s = src[i]; d = dst[i]; float a = ea[i];
                  m = (mode == 2) || (mode == 0 ? (a <= 0.f) : (a >= 0.f)); }
    else        { s = d = i - ne; m = true; }
    float e = -1e9f;
    if (m) {
        const float* ld = l + (size_t)d * C_DIM + h * 64;
        const float* rs = r + (size_t)s * C_DIM + h * 64;
        const float* at = att + h * 64;
        float acc = 0.f;
        #pragma unroll 8
        for (int c = 0; c < 64; ++c) {
            float hv = ld[c] + rs[c];
            float lr = hv > 0.f ? hv : 0.2f * hv;   // leaky_relu(0.2)
            acc += lr * at[c];
        }
        e = acc;
    }
    ebuf[(size_t)i * 2 + h] = e;
    atomicMax(&mx[d * 2 + h], f2ord(e));
}

__global__ void k_gat_sum(const float* __restrict__ ebuf, const float* __restrict__ r,
                          const int* __restrict__ src, const int* __restrict__ dst,
                          const int* __restrict__ mx, float* __restrict__ s,
                          float* __restrict__ numer, int ne, int n)
{
    int g = blockIdx.x * blockDim.x + threadIdx.x;
    int i = g >> 1, h = g & 1;
    if (i >= ne + n) return;
    float e = ebuf[(size_t)i * 2 + h];
    if (e <= -1e8f) return;                 // masked edge -> exp ~ 0
    int ss, d;
    if (i < ne) { ss = src[i]; d = dst[i]; } else { ss = d = i - ne; }
    float ex = expf(e - ord2f(mx[d * 2 + h]));
    atomicAdd(&s[d * 2 + h], ex);
    const float* rr = r + (size_t)ss * C_DIM + h * 64;
    float* nm = numer + (size_t)d * C_DIM + h * 64;
    #pragma unroll 8
    for (int c = 0; c < 64; ++c) atomicAdd(nm + c, ex * rr[c]);
}

__global__ void k_gat_div(const float* __restrict__ numer, const float* __restrict__ s,
                          const float* __restrict__ bias, float* __restrict__ out, int n)
{
    size_t i = (size_t)blockIdx.x * blockDim.x + threadIdx.x;
    if (i >= (size_t)n * C_DIM) return;
    int f = (int)(i & 127);
    size_t node = i >> 7;
    int h = f >> 6;
    out[i] = numer[i] / (s[node * 2 + h] + 1e-16f) + bias[f];
}

__global__ void k_fill(unsigned* __restrict__ p, unsigned v, size_t cnt)
{
    size_t i = (size_t)blockIdx.x * blockDim.x + threadIdx.x;
    if (i < cnt) p[i] = v;
}

// ---------------------------------------------------------------------------
extern "C" void kernel_launch(void* const* d_in, const int* in_sizes, int n_in,
                              void* d_out, int out_size, void* d_ws, size_t ws_size,
                              hipStream_t stream)
{
    (void)out_size; (void)ws_size;
    const float* x  = (const float*)d_in[0];
    const int*   ei = (const int*)d_in[1];
    const float* ea = (const float*)d_in[2];
    const int N = in_sizes[0] / 2048;
    const int E = in_sizes[2];
    const int* src = ei;
    const int* dst = ei + E;

    const float* P[64];
    for (int i = 3; i < n_in && i < 64; ++i) P[i] = (const float*)d_in[i];
    // 3:W011 4:b011 5:W012 6:b012 | 7-9:gn01 | 10-12:t11 13-15:gn11 | 16-18:t12
    // 19-21:gn12 | 22-24:t13 25-27:gn13 | 28:Wl 29:Wr 30:att 31:gat_bias
    // 32-34:gn21 | 35-37:t31 38-40:t32 41-43:t33

    const size_t nf = (size_t)N * C_DIM;
    float* ws   = (float*)d_ws;
    float* h0   = ws;                                 // [N,128]
    float* ybuf = h0 + nf;                            // [N,128]
    float* agg  = ybuf + nf;                          // [N,128] (also GAT numerator)
    float* lbuf = agg + nf;                           // [N,128] (also GAT output)
    float* rbuf = lbuf + nf;                          // [N,128]
    float* ebuf = rbuf + nf;                          // [(E+N)*2]
    float* degb = ebuf + (size_t)(E + N) * 2;         // [N]
    int*   mxb  = (int*)(degb + N);                   // [N*2]
    float* sb   = (float*)(mxb + (size_t)N * 2);      // [N*2]
    float* stats = sb + (size_t)N * 2;                // [256] sum|sumsq
    float* coef  = stats + 256;                       // [256] a|c

    const dim3 blk(256);
    const int gemGrid = (N + 127) / 128;
    auto grid1 = [](size_t c) { return dim3((unsigned)((c + 255) / 256)); };

    union { float f; unsigned u; } cu; cu.f = -1e9f;
    const unsigned encNeg = ((int)cu.u >= 0) ? cu.u : (cu.u ^ 0x7fffffffu);

    auto graphnorm_gelu = [&](float* buf, const float* g, const float* b, const float* al) {
        k_fill<<<grid1(256), blk, 0, stream>>>((unsigned*)stats, 0u, 256);
        k_colstats<<<dim3((N + 63) / 64), blk, 0, stream>>>(buf, stats, N);
        k_gncoef<<<dim3(1), dim3(128), 0, stream>>>(stats, g, b, al, coef, N);
        k_gngelu<<<grid1(nf), blk, 0, stream>>>(buf, coef, nf);
    };

    // ---- stem: h0 = gelu(gn01(x@[W011;W012] + b)) ----
    k_inproj<<<dim3(gemGrid), blk, 0, stream>>>(x, P[3], P[5], P[4], P[6], h0, N);
    graphnorm_gelu(h0, P[7], P[8], P[9]);

    const int t1i[3]  = {10, 16, 22};
    const int gn1i[3] = {13, 19, 25};
    const int t3i[3]  = {35, 38, 41};
    const size_t tot2 = (size_t)(E + N) * 2;

    for (int br = 0; br < 3; ++br) {            // mode: 0 = ea<=0, 1 = ea>=0, 2 = all
        // thres_block 1
        k_fill<<<grid1(nf), blk, 0, stream>>>((unsigned*)agg, 0u, nf);
        k_fill<<<grid1(N), blk, 0, stream>>>((unsigned*)degb, 0u, (size_t)N);
        k_edge_agg<<<grid1((size_t)E * 4), blk, 0, stream>>>(h0, src, dst, ea, br, agg, degb, E);
        k_gemm128<<<dim3(gemGrid), blk, 0, stream>>>(h0, P[t1i[br]], agg, P[t1i[br] + 1],
                                                     degb, P[t1i[br] + 2], ybuf, N);
        graphnorm_gelu(ybuf, P[gn1i[br]], P[gn1i[br] + 1], P[gn1i[br] + 2]);

        // GATv2
        k_gemm128<<<dim3(gemGrid), blk, 0, stream>>>(ybuf, P[28], nullptr, nullptr, nullptr, nullptr, lbuf, N);
        k_gemm128<<<dim3(gemGrid), blk, 0, stream>>>(ybuf, P[29], nullptr, nullptr, nullptr, nullptr, rbuf, N);
        k_fill<<<grid1((size_t)N * 2), blk, 0, stream>>>((unsigned*)mxb, encNeg, (size_t)N * 2);
        k_fill<<<grid1((size_t)N * 2), blk, 0, stream>>>((unsigned*)sb, 0u, (size_t)N * 2);
        k_fill<<<grid1(nf), blk, 0, stream>>>((unsigned*)agg, 0u, nf);
        k_gat_e  <<<grid1(tot2), blk, 0, stream>>>(lbuf, rbuf, src, dst, ea, br, P[30], ebuf, mxb, E, N);
        k_gat_sum<<<grid1(tot2), blk, 0, stream>>>(ebuf, rbuf, src, dst, mxb, sb, agg, E, N);
        k_gat_div<<<grid1(nf), blk, 0, stream>>>(agg, sb, P[31], lbuf, N);
        graphnorm_gelu(lbuf, P[32], P[33], P[34]);

        // thres_block 3 -> output slice
        k_fill<<<grid1(nf), blk, 0, stream>>>((unsigned*)agg, 0u, nf);
        k_fill<<<grid1(N), blk, 0, stream>>>((unsigned*)degb, 0u, (size_t)N);
        k_edge_agg<<<grid1((size_t)E * 4), blk, 0, stream>>>(lbuf, src, dst, ea, br, agg, degb, E);
        float* outb = (float*)d_out + (size_t)br * nf;
        k_gemm128<<<dim3(gemGrid), blk, 0, stream>>>(lbuf, P[t3i[br]], agg, P[t3i[br] + 1],
                                                     degb, P[t3i[br] + 2], outb, N);
    }
}
